// GCRN_44659069944064
// MI455X (gfx1250) — matrix operand order repaired and checked by reading, not compiled, not run on hardware
//
#include <hip/hip_runtime.h>
#include <hip/hip_bf16.h>
#include <stdint.h>

// Problem constants (fixed by the reference)
#define TT      8
#define NNODES  50000
#define NEDGES  800000
#define CDIM    64
#define MROWS   (TT * NNODES)     // 400000 rows, 25000 16-row tiles
#define NWAVES_TOT 5000           // 625 blocks * 8 waves
#define TILES_PER_WAVE 5          // 25000 / 5000

typedef __attribute__((ext_vector_type(16))) __bf16 v16bf;
typedef __attribute__((ext_vector_type(8)))  __bf16 v8bf;
typedef __attribute__((ext_vector_type(2)))  __bf16 v2bf;
typedef __attribute__((ext_vector_type(8)))  float  v8f;

#if defined(__has_builtin)
#if __has_builtin(__builtin_amdgcn_cvt_pk_bf16_f32)
#define HAVE_PK_BF16 1
#endif
#endif

static __device__ __forceinline__ unsigned short f2bfu(float f) {
    unsigned u = __float_as_uint(f);
    u += 0x7FFFu + ((u >> 16) & 1u);           // round-to-nearest-even
    return (unsigned short)(u >> 16);
}

static __device__ __forceinline__ unsigned pk2bf(float lo, float hi) {
#ifdef HAVE_PK_BF16
    v2bf r = __builtin_amdgcn_cvt_pk_bf16_f32(lo, hi);
    unsigned u;
    __builtin_memcpy(&u, &r, 4);
    return u;
#else
    return (unsigned)f2bfu(lo) | ((unsigned)f2bfu(hi) << 16);
#endif
}

static __device__ __forceinline__ float atomAddF(float* p, float v) {
    return __hip_atomic_fetch_add(p, v, __ATOMIC_RELAXED, __HIP_MEMORY_SCOPE_AGENT);
}

// ---------------- degree / normalization ----------------
__global__ void k_deg(const int* __restrict__ ei, const float* __restrict__ w,
                      float* __restrict__ deg) {
    int e = blockIdx.x * blockDim.x + threadIdx.x;
    if (e < NEDGES) atomAddF(&deg[ei[e]], w[e]);
}

__global__ void k_dinv(float* __restrict__ deg) {   // in-place deg -> dinv
    int i = blockIdx.x * blockDim.x + threadIdx.x;
    if (i < NNODES) {
        float d = deg[i];
        deg[i] = (d > 0.f) ? rsqrtf(fmaxf(d, 1e-12f)) : 0.f;
    }
}

__global__ void k_nw(const int* __restrict__ ei, const float* __restrict__ w,
                     const float* __restrict__ dinv, float* __restrict__ nw) {
    int e = blockIdx.x * blockDim.x + threadIdx.x;
    if (e < NEDGES) {
        int s = ei[e], d = ei[NEDGES + e];
        nw[e] = -w[e] * dinv[s] * dinv[d];
    }
}

// ---------------- weight prep: f32 [k][n] -> bf16 [n][k], fused biases ----------------
__global__ void k_prep(const float* __restrict__ Wz0, const float* __restrict__ Wz1,
                       const float* __restrict__ Wh0, const float* __restrict__ Wh1,
                       const float* __restrict__ bxz, const float* __restrict__ bhz,
                       const float* __restrict__ bxh, const float* __restrict__ bhh,
                       unsigned short* __restrict__ Wt, float* __restrict__ bz,
                       float* __restrict__ bh) {
    const float* Ws[4] = {Wz0, Wz1, Wh0, Wh1};
    for (int m = 0; m < 4; ++m) {
        for (int i = threadIdx.x; i < 4096; i += blockDim.x) {
            int n = i >> 6, k = i & 63;
            Wt[m * 4096 + i] = f2bfu(Ws[m][k * 64 + n]);   // Wt[m][n][k]
        }
    }
    for (int i = threadIdx.x; i < 64; i += blockDim.x) {
        bz[i] = bxz[i] + bhz[i];
        bh[i] = bxh[i] + bhh[i];
    }
}

// ---------------- graph scatter: Lx[t,dst,c] += nw * x[t,src,c] ----------------
__global__ void k_scatter(const int* __restrict__ ei, const float* __restrict__ nw,
                          const float* __restrict__ x, float* __restrict__ Lx) {
    long tid = (long)blockIdx.x * blockDim.x + threadIdx.x;
    if (tid >= (long)NEDGES * CDIM) return;
    int e = (int)(tid >> 6);
    int c = (int)(tid & 63);
    int s = ei[e], d = ei[NEDGES + e];
    float wv = nw[e];
#pragma unroll
    for (int t = 0; t < TT; ++t) {
        float xv = x[((size_t)t * NNODES + s) * CDIM + c];
        atomAddF(&Lx[((size_t)t * NNODES + d) * CDIM + c], wv * xv);
    }
}

// ---------------- bulk f32 -> bf16 conversion (bandwidth-bound pass) ----------------
__global__ void k_cvt(const float* __restrict__ in, uint4* __restrict__ outp, long n8) {
    long i = (long)blockIdx.x * blockDim.x + threadIdx.x;   // one uint4 = 8 bf16
    if (i >= n8) return;
    const float4* p = (const float4*)in + i * 2;
    float4 a = p[0], b = p[1];
    uint4 o;
    o.x = pk2bf(a.x, a.y);
    o.y = pk2bf(a.z, a.w);
    o.z = pk2bf(b.x, b.y);
    o.w = pk2bf(b.z, b.w);
    outp[i] = o;
}

// ---------------- fused WMMA kernel ----------------
static __device__ __forceinline__ v16bf loadA(const unsigned short* p, int k0) {
    v8bf lo = *(const v8bf*)(p + k0);
    v8bf hi = *(const v8bf*)(p + k0 + 8);
    return __builtin_shufflevector(lo, hi,
        0, 1, 2, 3, 4, 5, 6, 7, 8, 9, 10, 11, 12, 13, 14, 15);
}

__global__ __launch_bounds__(256) void k_gcrn(
        const unsigned short* __restrict__ xb, const unsigned short* __restrict__ lb,
        const unsigned short* __restrict__ Wt,
        const float* __restrict__ bz, const float* __restrict__ bh,
        const float* __restrict__ wlin, const float* __restrict__ blin,
        float* __restrict__ out) {
    __shared__ __align__(32) unsigned short sW[4 * 4096];  // 32 KB: 4 mats [n][k] bf16
    __shared__ float sBz[64], sBh[64], sWl[64];

    {   // cooperative stage of weights into LDS (once per block, amortized over 5 tiles/wave)
        const uint4* s = (const uint4*)Wt;
        uint4* d = (uint4*)sW;
        for (int i = threadIdx.x; i < 2048; i += 256) d[i] = s[i];
        for (int i = threadIdx.x; i < 64; i += 256) {
            sBz[i] = bz[i]; sBh[i] = bh[i]; sWl[i] = wlin[i];
        }
    }
    __syncthreads();

    const int wave = threadIdx.x >> 5;
    const int lane = threadIdx.x & 31;
    const int lm = lane & 15;       // row-in-tile (A) / column (B,C,D)
    const int lh = lane >> 4;       // which K/M half this lane covers
    const int basetile = blockIdx.x * 8 + wave;
    const float bl = blin[0];

#pragma unroll 1
    for (int it = 0; it < TILES_PER_WAVE; ++it) {
        const size_t rowbase = (size_t)(basetile + it * NWAVES_TOT) * 16;
        const unsigned short* xr = xb + (rowbase + lm) * CDIM;
        const unsigned short* lr = lb + (rowbase + lm) * CDIM;

        // Issue ALL A-operand loads for this tile up front (8x global_load_b128)
        // so the scheduler can use partial s_wait_loadcnt and overlap loads with
        // earlier WMMA groups.  A 16x32 layout (matches B): lanes 0-15 hold
        // K=[k0..k0+15]; lanes 16-31 K=[k0+16..k0+31].
        v16bf A[4];
        A[0] = loadA(xr, 0  + lh * 16);   // x,  ks=0
        A[1] = loadA(lr, 0  + lh * 16);   // Lx, ks=0
        A[2] = loadA(xr, 32 + lh * 16);   // x,  ks=1
        A[3] = loadA(lr, 32 + lh * 16);   // Lx, ks=1

        v8f accZ[4] = {};
        v8f accH[4] = {};

#pragma unroll
        for (int ks = 0; ks < 2; ++ks) {
            const int kb = lh * 16 + ks * 32;   // B: lane=column, contiguous 16-K chunk
            const v16bf ax = A[ks * 2 + 0];
            const v16bf al = A[ks * 2 + 1];
#pragma unroll
            for (int nt = 0; nt < 4; ++nt) {
                const int col = nt * 16 + lm;
                v16bf b;
                b = *(const v16bf*)(sW + 0 * 4096 + col * 64 + kb);
                accZ[nt] = __builtin_amdgcn_wmma_f32_16x16x32_bf16(
                    false, ax, false, b, (short)0, accZ[nt], false, false);
                b = *(const v16bf*)(sW + 1 * 4096 + col * 64 + kb);
                accZ[nt] = __builtin_amdgcn_wmma_f32_16x16x32_bf16(
                    false, al, false, b, (short)0, accZ[nt], false, false);
                b = *(const v16bf*)(sW + 2 * 4096 + col * 64 + kb);
                accH[nt] = __builtin_amdgcn_wmma_f32_16x16x32_bf16(
                    false, ax, false, b, (short)0, accH[nt], false, false);
                b = *(const v16bf*)(sW + 3 * 4096 + col * 64 + kb);
                accH[nt] = __builtin_amdgcn_wmma_f32_16x16x32_bf16(
                    false, al, false, b, (short)0, accH[nt], false, false);
            }
        }

        // Gate math on C/D layout: element r of lane (lm,lh) = (row r+lh*8, col nt*16+lm)
        float part[8] = {0.f, 0.f, 0.f, 0.f, 0.f, 0.f, 0.f, 0.f};
#pragma unroll
        for (int nt = 0; nt < 4; ++nt) {
            const int col = nt * 16 + lm;
            const float bzv = sBz[col], bhv = sBh[col], wl = sWl[col];
#pragma unroll
            for (int r = 0; r < 8; ++r) {
                float az = accZ[nt][r] + bzv;
                float ah = accH[nt][r] + bhv;
                float z  = 1.f / (1.f + __expf(-az));         // sigmoid
                float e2 = __expf(-2.f * ah);                 // tanh via exp
                float ht = (1.f - e2) / (1.f + e2);
                float hv = (1.f - z) * ht;                    // H = (1-Z)*Ht (H0 = 0)
                hv = fmaxf(hv, 0.f);                          // relu
                part[r] += hv * wl;                           // partial of h @ W_lin
            }
        }
        // reduce the 64 columns: sum across the 16 lanes of each half
#pragma unroll
        for (int m = 1; m < 16; m <<= 1) {
#pragma unroll
            for (int r = 0; r < 8; ++r) part[r] += __shfl_xor(part[r], m, 32);
        }
        if (lm == 0) {
#pragma unroll
            for (int r = 0; r < 8; ++r)
                out[rowbase + lh * 8 + r] = part[r] + bl;
        }
    }
}

// ---------------- host entry ----------------
extern "C" void kernel_launch(void* const* d_in, const int* in_sizes, int n_in,
                              void* d_out, int out_size, void* d_ws, size_t ws_size,
                              hipStream_t stream) {
    const float* x    = (const float*)d_in[0];
    const int*   ei   = (const int*)d_in[1];
    const float* ew   = (const float*)d_in[2];
    const float* Wxz0 = (const float*)d_in[3];
    const float* Wxz1 = (const float*)d_in[4];
    const float* bxz  = (const float*)d_in[5];
    const float* bhz  = (const float*)d_in[8];
    const float* Wxh0 = (const float*)d_in[15];
    const float* Wxh1 = (const float*)d_in[16];
    const float* bxh  = (const float*)d_in[17];
    const float* bhh  = (const float*)d_in[20];
    const float* wlin = (const float*)d_in[21];
    const float* blin = (const float*)d_in[22];

    char* ws = (char*)d_ws;
    size_t off = 0;
    auto take = [&](size_t bytes) -> char* {
        char* p = ws + off;
        off = (off + bytes + 255) & ~(size_t)255;
        return p;
    };
    float*          deg = (float*)take((size_t)NNODES * 4);
    float*          nw  = (float*)take((size_t)NEDGES * 4);
    float*          Lx  = (float*)take((size_t)MROWS * CDIM * 4);
    unsigned short* xb  = (unsigned short*)take((size_t)MROWS * CDIM * 2);
    unsigned short* lb  = (unsigned short*)take((size_t)MROWS * CDIM * 2);
    unsigned short* Wt  = (unsigned short*)take(4 * 4096 * 2);
    float*          bz  = (float*)take(64 * 4);
    float*          bh  = (float*)take(64 * 4);

    hipMemsetAsync(deg, 0, (size_t)NNODES * 4, stream);
    hipMemsetAsync(Lx,  0, (size_t)MROWS * CDIM * 4, stream);

    const long n8 = (long)MROWS * CDIM / 8;   // 3,200,000 uint4 groups

    k_deg <<<(NEDGES + 255) / 256, 256, 0, stream>>>(ei, ew, deg);
    k_dinv<<<(NNODES + 255) / 256, 256, 0, stream>>>(deg);
    k_nw  <<<(NEDGES + 255) / 256, 256, 0, stream>>>(ei, ew, deg, nw);
    k_prep<<<1, 256, 0, stream>>>(Wxz0, Wxz1, Wxh0, Wxh1, bxz, bhz, bxh, bhh, Wt, bz, bh);
    k_cvt <<<(n8 + 255) / 256, 256, 0, stream>>>(x, (uint4*)xb, n8);
    k_scatter<<<((long)NEDGES * CDIM) / 256, 256, 0, stream>>>(ei, nw, x, Lx);
    k_cvt <<<(n8 + 255) / 256, 256, 0, stream>>>(Lx, (uint4*)lb, n8);
    k_gcrn<<<NWAVES_TOT / 8, 256, 0, stream>>>(xb, lb, Wt, bz, bh, wlin, blin, (float*)d_out);
}
